// SparseExpertAttention_74577812127850
// MI455X (gfx1250) — matrix operand order repaired and checked by reading, not compiled, not run on hardware
//
#include <hip/hip_runtime.h>
#include <hip/hip_bf16.h>

typedef __attribute__((ext_vector_type(16))) _Float16 v16h;
typedef __attribute__((ext_vector_type(8)))  _Float16 v8h;
typedef __attribute__((ext_vector_type(8)))  float    v8f;

#define NB   8
#define NT   2048
#define NC   768
#define NE   3
#define NHD  256
#define NTOK (NB*NT)      // 16384
#define NQKV (3*NC)       // 2304: q[0:768) k[768:1536) v[1536:2304)
#define ATT_SCALE 0.0625f // 1/sqrt(256)
#define NEGBIG (-1e30f)

// ---------------- WMMA helpers (CDNA5 gfx1250, wave32) ----------------

static __device__ __forceinline__ v8f zero8() {
  v8f z;
#pragma unroll
  for (int i = 0; i < 8; ++i) z[i] = 0.0f;
  return z;
}

// A-matrix 16x32 f16 fragment from row-major memory (ISA 7.12.2 layout):
// lane L: row = L&15 ; halves0..7 = K[(L>>4)*8 .. +7] ; halves8..15 = +16.
static __device__ __forceinline__ v16h load_a16(const _Float16* s, int ld, int lane) {
  const int m  = lane & 15;
  const int ko = (lane >> 4) * 8;
  v8h lo = *(const v8h*)(s + (size_t)m * ld + ko);
  v8h hi = *(const v8h*)(s + (size_t)m * ld + ko + 16);
  return __builtin_shufflevector(lo, hi, 0,1,2,3,4,5,6,7,8,9,10,11,12,13,14,15);
}

// B-matrix 32x16 f16 fragment from row-major (K x N) memory:
// lane L: col N = L&15 ; halves0..7 = K[(L>>4)*8 .. +7] ; halves8..15 = +16.
static __device__ __forceinline__ v16h load_b16(const _Float16* s, int ld, int lane) {
  const int nn = lane & 15;
  const int ko = (lane >> 4) * 8;
  v16h r;
#pragma unroll
  for (int i = 0; i < 8; ++i) {
    r[i]     = s[(size_t)(ko + i) * ld + nn];
    r[8 + i] = s[(size_t)(ko + 16 + i) * ld + nn];
  }
  return r;
}

static __device__ __forceinline__ v8f wmma16(v16h a, v16h b, v8f c) {
  return __builtin_amdgcn_wmma_f32_16x16x32_f16(false, a, false, b, (short)0, c, false, false);
}

// ---------------- small prep kernels ----------------

__global__ void k_cvt_f16(const float* __restrict__ src, _Float16* __restrict__ dst, int n) {
  int i = blockIdx.x * blockDim.x + threadIdx.x;
  if (i < n) dst[i] = (_Float16)src[i];
}

// wh[r][c]: c<1536 -> Wqk[r][c] ; else Wv[e][r][hd], e=(c-1536)/256
__global__ void k_pack_wqkv(const float* __restrict__ Wqk, const float* __restrict__ Wv,
                            _Float16* __restrict__ wh) {
  int i = blockIdx.x * blockDim.x + threadIdx.x;
  if (i >= NC * NQKV) return;
  int r = i / NQKV, c = i % NQKV;
  float v;
  if (c < 2 * NC) v = Wqk[(size_t)r * (2 * NC) + c];
  else {
    int cc = c - 2 * NC, e = cc >> 8, hd = cc & 255;
    v = Wv[((size_t)e * NC + r) * NHD + hd];
  }
  wh[i] = (_Float16)v;
}

__global__ void k_pack_bias(const float* __restrict__ bqk, const float* __restrict__ bv,
                            float* __restrict__ bias) {
  int i = blockIdx.x * blockDim.x + threadIdx.x;
  if (i >= NQKV) return;
  bias[i] = (i < 2 * NC) ? bqk[i] : bv[i - 2 * NC];
}

// ---------------- gating: w_e[token][e] (top-2-of-3, re-softmaxed) ----------------
// one wave (32 lanes) per token: lane j owns hidden unit j of 32
__global__ void k_gate(const float* __restrict__ x,
                       const float* __restrict__ Wg1, const float* __restrict__ bg1,
                       const float* __restrict__ Wg2, const float* __restrict__ bg2,
                       float* __restrict__ we) {
  const int tok  = blockIdx.x * 8 + (threadIdx.x >> 5);
  const int lane = threadIdx.x & 31;
  const float* xr = x + (size_t)tok * NC;
  float h = bg1[lane];
  for (int c = 0; c < NC; ++c) h += xr[c] * Wg1[c * 32 + lane];
  h = fmaxf(h, 0.0f);
  float g0 = h * Wg2[lane * NE + 0];
  float g1 = h * Wg2[lane * NE + 1];
  float g2 = h * Wg2[lane * NE + 2];
#pragma unroll
  for (int m = 16; m >= 1; m >>= 1) {
    g0 += __shfl_xor(g0, m, 32);
    g1 += __shfl_xor(g1, m, 32);
    g2 += __shfl_xor(g2, m, 32);
  }
  if (lane == 0) {
    float g[NE] = {g0 + bg2[0], g1 + bg2[1], g2 + bg2[2]};
    float mx = fmaxf(g[0], fmaxf(g[1], g[2]));
    float p[NE], s = 0.0f;
#pragma unroll
    for (int e = 0; e < NE; ++e) { p[e] = __expf(g[e] - mx); s += p[e]; }
#pragma unroll
    for (int e = 0; e < NE; ++e) p[e] /= s;
    int i0 = 0;
    if (p[1] > p[i0]) i0 = 1;
    if (p[2] > p[i0]) i0 = 2;
    int i1 = (i0 == 0) ? 1 : 0;
#pragma unroll
    for (int e = 0; e < NE; ++e)
      if (e != i0 && p[e] > p[i1]) i1 = e;
    float w0 = 1.0f / (1.0f + __expf(p[i1] - p[i0]));
    float o[NE] = {0.0f, 0.0f, 0.0f};
    o[i0] = w0; o[i1] = 1.0f - w0;
#pragma unroll
    for (int e = 0; e < NE; ++e) we[tok * NE + e] = o[e];
  }
}

// ---------------- fused qkv projection GEMM (16384 x 2304 x 768), f16 WMMA ----------
// block = 4 waves; wave computes a 16x64 tile
__global__ void k_gemm_qkv(const _Float16* __restrict__ xh, const _Float16* __restrict__ wh,
                           const float* __restrict__ bias, _Float16* __restrict__ qkvh) {
  const int lane = threadIdx.x & 31;
  const int wv   = threadIdx.x >> 5;
  const int row0 = blockIdx.y * 64 + wv * 16;
  const int col0 = blockIdx.x * 64;
  v8f acc[4];
#pragma unroll
  for (int j = 0; j < 4; ++j) acc[j] = zero8();
  for (int k0 = 0; k0 < NC; k0 += 32) {
    v16h a = load_a16(xh + (size_t)row0 * NC + k0, NC, lane);
#pragma unroll
    for (int j = 0; j < 4; ++j) {
      v16h b = load_b16(wh + (size_t)k0 * NQKV + col0 + j * 16, NQKV, lane);
      acc[j] = wmma16(a, b, acc[j]);
    }
  }
  const int n = lane & 15, mo = (lane >> 4) * 8;
#pragma unroll
  for (int j = 0; j < 4; ++j) {
    const int col = col0 + j * 16 + n;
    const float bz = bias[col];
#pragma unroll
    for (int r = 0; r < 8; ++r)
      qkvh[(size_t)(row0 + r + mo) * NQKV + col] = (_Float16)(acc[j][r] + bz);
  }
}

// ---------------- depthwise conv (k=3, pad=1) over time on expert-0 q ----------------
__global__ void k_conv_q0(const _Float16* __restrict__ qkvh, const float* __restrict__ cw,
                          const float* __restrict__ cb, _Float16* __restrict__ q0c) {
  int i = blockIdx.x * blockDim.x + threadIdx.x; // < NB*NT*NHD
  int d = i & 255, t = (i >> 8) & (NT - 1), b = i >> 19;
  const size_t row = (size_t)(b * NT + t) * NQKV + d; // q0 lives in cols [0,256)
  float acc = cb[d];
  float w0 = cw[d * 3 + 0], w1 = cw[d * 3 + 1], w2 = cw[d * 3 + 2];
  if (t > 0)      acc += w0 * (float)qkvh[row - NQKV];
  acc += w1 * (float)qkvh[row];
  if (t < NT - 1) acc += w2 * (float)qkvh[row + NQKV];
  q0c[(size_t)(b * NT + t) * NHD + d] = (_Float16)acc;
}

// ---------------- K transpose: kT[(b*3+e)*256+hd][t] = k[b][t][e*256+hd] ----------------
__global__ void k_trans_k(const _Float16* __restrict__ qkvh, _Float16* __restrict__ kTb) {
  __shared__ _Float16 tile[32][33];
  const int tb = blockIdx.x * 32, hb = blockIdx.y * 32, be = blockIdx.z;
  const int b = be / NE, e = be % NE;
#pragma unroll
  for (int i = 0; i < 4; ++i) {
    int t = tb + threadIdx.y + i * 8, hd = hb + threadIdx.x;
    tile[threadIdx.y + i * 8][threadIdx.x] =
        qkvh[(size_t)(b * NT + t) * NQKV + NC + e * NHD + hd];
  }
  __syncthreads();
#pragma unroll
  for (int i = 0; i < 4; ++i) {
    int hd = hb + threadIdx.y + i * 8, t = tb + threadIdx.x;
    kTb[(size_t)((b * NE + e) * NHD + hd) * NT + t] = tile[threadIdx.x][threadIdx.y + i * 8];
  }
}

// ---------------- flash attention per (query-tile, batch, expert) ----------------
// block = 4 waves, each wave owns 16 query rows; streams 32 keys/iter (2 WMMA N-tiles)
__global__ void __launch_bounds__(128) k_attn(
    const _Float16* __restrict__ qkvh, const _Float16* __restrict__ q0c,
    const _Float16* __restrict__ kTb, const float* __restrict__ we,
    _Float16* __restrict__ ch) {
  const int e = blockIdx.z, b = blockIdx.y;
  const int wv = threadIdx.x >> 5, lane = threadIdx.x & 31;
  const int qt = blockIdx.x * 64 + wv * 16;

  const _Float16* qptr; int qld;
  if (e == 0) { qptr = q0c + (size_t)(b * NT + qt) * NHD;            qld = NHD;  }
  else        { qptr = qkvh + (size_t)(b * NT + qt) * NQKV + e * NHD; qld = NQKV; }

  v16h qa[8];
#pragma unroll
  for (int c = 0; c < 8; ++c) qa[c] = load_a16(qptr + c * 32, qld, lane);

  v8f o[16];
#pragma unroll
  for (int j = 0; j < 16; ++j) o[j] = zero8();
  float mrow[8], lrow[8];
#pragma unroll
  for (int r = 0; r < 8; ++r) { mrow[r] = NEGBIG; lrow[r] = 0.0f; }

  const _Float16* kbase = kTb + (size_t)((b * NE + e) * NHD) * NT;          // [hd][t]
  const _Float16* vbase = qkvh + (size_t)(b * NT) * NQKV + 2 * NC + e * NHD; // [t][hd]

  __shared__ __align__(16) _Float16 pT[4][16 * 32];
  _Float16* myP = pT[wv];

  const int n  = lane & 15;
  const int mo = (lane >> 4) * 8;
  const int qend = qt + 15;

  for (int kt = 0; kt <= qend; kt += 32) {
    // ---- S = Q K^T over two 16-key sub-tiles ----
    v8f s0 = zero8(), s1 = zero8();
#pragma unroll
    for (int c = 0; c < 8; ++c) {
      const _Float16* krow = kbase + (size_t)(c * 32) * NT + kt;
      s0 = wmma16(qa[c], load_b16(krow,      NT, lane), s0);
      s1 = wmma16(qa[c], load_b16(krow + 16, NT, lane), s1);
    }
    if (kt + 32 <= qend) // prefetch next K tile into caches
      __builtin_prefetch(kbase + (size_t)lane * NT + kt + 32, 0, 0);

    // ---- scale + causal mask + online softmax (row = r+mo, col group = n) ----
    float alpha[8];
#pragma unroll
    for (int r = 0; r < 8; ++r) {
      const int qrow = qt + r + mo;
      float a0 = s0[r] * ATT_SCALE; if (kt + n      > qrow) a0 = NEGBIG;
      float a1 = s1[r] * ATT_SCALE; if (kt + 16 + n > qrow) a1 = NEGBIG;
      float mx = fmaxf(a0, a1);
      mx = fmaxf(mx, __shfl_xor(mx, 1, 32));
      mx = fmaxf(mx, __shfl_xor(mx, 2, 32));
      mx = fmaxf(mx, __shfl_xor(mx, 4, 32));
      mx = fmaxf(mx, __shfl_xor(mx, 8, 32));
      const float mnew = fmaxf(mrow[r], mx);
      alpha[r] = __expf(mrow[r] - mnew);
      const float p0 = __expf(a0 - mnew);
      const float p1 = __expf(a1 - mnew);
      float rs = p0 + p1;
      rs += __shfl_xor(rs, 1, 32);
      rs += __shfl_xor(rs, 2, 32);
      rs += __shfl_xor(rs, 4, 32);
      rs += __shfl_xor(rs, 8, 32);
      lrow[r] = lrow[r] * alpha[r] + rs;
      mrow[r] = mnew;
      myP[(r + mo) * 32 + n]      = (_Float16)p0;
      myP[(r + mo) * 32 + 16 + n] = (_Float16)p1;
    }
    // same-wave DS ops are in-order: drain stores before re-reading as A-fragment
    asm volatile("s_wait_dscnt 0" ::: "memory");
    v16h pa = load_a16(myP, 32, lane);

    // ---- O = diag(alpha)*O + P V ----
#pragma unroll
    for (int j = 0; j < 16; ++j) {
      v8f oj = o[j];
#pragma unroll
      for (int r = 0; r < 8; ++r) oj[r] *= alpha[r];
      v16h vb = load_b16(vbase + (size_t)kt * NQKV + j * 16, NQKV, lane);
      o[j] = wmma16(pa, vb, oj);
    }
  }

  // ---- epilogue: O / l * w_e -> combined f16 slice [e*256,(e+1)*256) ----
  float fac[8];
#pragma unroll
  for (int r = 0; r < 8; ++r) {
    const int row = qt + r + mo;
    fac[r] = we[(b * NT + row) * NE + e] / lrow[r];
  }
#pragma unroll
  for (int j = 0; j < 16; ++j)
#pragma unroll
    for (int r = 0; r < 8; ++r)
      ch[(size_t)(b * NT + qt + r + mo) * NC + e * NHD + j * 16 + n] =
          (_Float16)(o[j][r] * fac[r]);
}

// ---------------- output projection GEMM (16384 x 768 x 768), f32 out ----------------
__global__ void k_gemm_out(const _Float16* __restrict__ chh, const _Float16* __restrict__ wph,
                           const float* __restrict__ bp, float* __restrict__ out) {
  const int lane = threadIdx.x & 31;
  const int wv   = threadIdx.x >> 5;
  const int row0 = blockIdx.y * 64 + wv * 16;
  const int col0 = blockIdx.x * 64;
  v8f acc[4];
#pragma unroll
  for (int j = 0; j < 4; ++j) acc[j] = zero8();
  for (int k0 = 0; k0 < NC; k0 += 32) {
    v16h a = load_a16(chh + (size_t)row0 * NC + k0, NC, lane);
#pragma unroll
    for (int j = 0; j < 4; ++j) {
      v16h b = load_b16(wph + (size_t)k0 * NC + col0 + j * 16, NC, lane);
      acc[j] = wmma16(a, b, acc[j]);
    }
  }
  const int n = lane & 15, mo = (lane >> 4) * 8;
#pragma unroll
  for (int j = 0; j < 4; ++j) {
    const int col = col0 + j * 16 + n;
    const float bz = bp[col];
#pragma unroll
    for (int r = 0; r < 8; ++r)
      out[(size_t)(row0 + r + mo) * NC + col] = acc[j][r] + bz;
  }
}

// ---------------- launcher ----------------

extern "C" void kernel_launch(void* const* d_in, const int* in_sizes, int n_in,
                              void* d_out, int out_size, void* d_ws, size_t ws_size,
                              hipStream_t stream) {
  (void)in_sizes; (void)n_in; (void)out_size; (void)ws_size;
  const float* x      = (const float*)d_in[0];
  const float* Wqk    = (const float*)d_in[1];
  const float* bqk    = (const float*)d_in[2];
  const float* Wv     = (const float*)d_in[3];
  const float* bv     = (const float*)d_in[4];
  const float* conv_w = (const float*)d_in[5];
  const float* conv_b = (const float*)d_in[6];
  // d_in[7] = glob   : fully causally masked -> provably no contribution
  // d_in[8] = pos_bias: only applied when T<=64; T=2048 -> unused
  const float* Wg1    = (const float*)d_in[9];
  const float* bg1    = (const float*)d_in[10];
  const float* Wg2    = (const float*)d_in[11];
  const float* bg2    = (const float*)d_in[12];
  const float* Wp     = (const float*)d_in[13];
  const float* bp     = (const float*)d_in[14];
  float* out = (float*)d_out;

  char* w = (char*)d_ws;
  constexpr size_t oXH  = 0;                                   // x f16: 16384x768
  constexpr size_t oWH  = oXH  + (size_t)NTOK * NC * 2;        // packed Wqkv f16: 768x2304
  constexpr size_t oWP  = oWH  + (size_t)NC * NQKV * 2;        // Wp f16: 768x768
  constexpr size_t oBQ  = oWP  + (size_t)NC * NC * 2;          // bias qkv f32: 2304
  constexpr size_t oQKV = oBQ  + (size_t)NQKV * 4;             // qkv f16: 16384x2304
  constexpr size_t oQ0  = oQKV + (size_t)NTOK * NQKV * 2;      // conv'd q0 f16: 16384x256
  constexpr size_t oKT  = oQ0  + (size_t)NTOK * NHD * 2;       // k transposed f16: 24x256x2048
  constexpr size_t oWE  = oKT  + (size_t)NB * NE * NHD * NT * 2; // w_e f32: 16384x3
  constexpr size_t oCH  = oWE  + (size_t)NTOK * NE * 4;        // combined f16: 16384x768
  _Float16* xh   = (_Float16*)(w + oXH);
  _Float16* wh   = (_Float16*)(w + oWH);
  _Float16* wph  = (_Float16*)(w + oWP);
  float*    bqkv = (float*)   (w + oBQ);
  _Float16* qkvh = (_Float16*)(w + oQKV);
  _Float16* q0c  = (_Float16*)(w + oQ0);
  _Float16* kTb  = (_Float16*)(w + oKT);
  float*    weW  = (float*)   (w + oWE);
  _Float16* chh  = (_Float16*)(w + oCH);

  k_cvt_f16  <<<(NTOK * NC) / 256, 256, 0, stream>>>(x, xh, NTOK * NC);
  k_pack_wqkv<<<(NC * NQKV) / 256, 256, 0, stream>>>(Wqk, Wv, wh);
  k_cvt_f16  <<<(NC * NC) / 256, 256, 0, stream>>>(Wp, wph, NC * NC);
  k_pack_bias<<<(NQKV + 255) / 256, 256, 0, stream>>>(bqk, bv, bqkv);
  k_gate     <<<NTOK / 8, 256, 0, stream>>>(x, Wg1, bg1, Wg2, bg2, weW);
  k_gemm_qkv <<<dim3(NQKV / 64, NTOK / 64), 128, 0, stream>>>(xh, wh, bqkv, qkvh);
  k_conv_q0  <<<(NB * NT * NHD) / 256, 256, 0, stream>>>(qkvh, conv_w, conv_b, q0c);
  k_trans_k  <<<dim3(NT / 32, NHD / 32, NB * NE), dim3(32, 8), 0, stream>>>(qkvh, kTb);
  k_attn     <<<dim3(NT / 64, NB, NE), 128, 0, stream>>>(qkvh, q0c, kTb, weW, chh);
  k_gemm_out <<<dim3(NC / 64, NTOK / 64), 128, 0, stream>>>(chh, wph, bp, out);
}
